// FlashPhiAttention_53755810676906
// MI455X (gfx1250) — matrix operand order repaired and checked by reading, not compiled, not run on hardware
//
#include <hip/hip_runtime.h>
#include <hip/hip_bf16.h>

// ---------------------------------------------------------------------------
// Phi-style attention block for MI455X (gfx1250, wave32, WMMA + TDM).
//   B=4, S=1024, T=4096, H=32, D=80 (padded to 96), HID=2560, ROT=32.
// GEMMs/attention: v_wmma_f32_16x16x32_bf16. bf16 tile staging: Tensor Data
// Mover (tensor_load_to_lds, TENSORcnt, 6-arg builtin) with
// global_load_async_to_lds_b128 fallback. f32->bf16: +0x8000 + v_perm_b32.
// ---------------------------------------------------------------------------

#define T_TOK   4096
#define SEQ     1024
#define NB      4
#define NH      32
#define HD      80
#define DPAD    96
#define HID     2560
#define NQKV    7680
#define ROT     32
#define RHALF   16
#define SCALE   0.11180339887498949f   // 80^-0.5

typedef __bf16    v16bf __attribute__((ext_vector_type(16)));
typedef float     v8f   __attribute__((ext_vector_type(8)));
typedef unsigned  v8u   __attribute__((ext_vector_type(8)));
typedef unsigned  uv4   __attribute__((ext_vector_type(4)));
typedef int       iv4   __attribute__((ext_vector_type(4)));
typedef int       iv8   __attribute__((ext_vector_type(8)));

#if defined(__has_builtin)
#  if __has_builtin(__builtin_amdgcn_tensor_load_to_lds)
#    define HAVE_TDM 1
#  endif
#endif

// cheap f32->bf16 (round-half-away): 1 add
static __device__ __forceinline__ unsigned short f2bf(float f) {
    unsigned u = __builtin_bit_cast(unsigned, f) + 0x8000u;
    return (unsigned short)(u >> 16);
}
// pack two f32 -> packed bf16x2: 2 adds + 1 v_perm_b32
static __device__ __forceinline__ unsigned pack_bf16(float a, float b) {
    unsigned ua = __builtin_bit_cast(unsigned, a) + 0x8000u;
    unsigned ub = __builtin_bit_cast(unsigned, b) + 0x8000u;
    // src0=ub (bytes 4-7), src1=ua (bytes 0-3): take high halves of each
    return __builtin_amdgcn_perm(ub, ua, 0x07060302u);
}
static __device__ __forceinline__ float bf2f(unsigned short h) {
    return __builtin_bit_cast(float, (unsigned)h << 16);
}
static __device__ __forceinline__ v8u mk8(uint4 lo, uint4 hi) {
    v8u r;
    r[0] = lo.x; r[1] = lo.y; r[2] = lo.z; r[3] = lo.w;
    r[4] = hi.x; r[5] = hi.y; r[6] = hi.z; r[7] = hi.w;
    return r;
}
static __device__ __forceinline__ v8f wmma_bf16(v8u a, v8u b, v8f c) {
    return __builtin_amdgcn_wmma_f32_16x16x32_bf16(
        false, __builtin_bit_cast(v16bf, a),
        false, __builtin_bit_cast(v16bf, b),
        (short)0, c, false, false);
}

// -- CDNA5 async copy (fallback path): 16B global -> LDS, ASYNCcnt ----------
static __device__ __forceinline__ void async_b128(void* lds_ptr, const void* gptr) {
    unsigned lds_off = (unsigned)(unsigned long long)lds_ptr;   // flat LDS addr[31:0]
    asm volatile("global_load_async_to_lds_b128 %0, %1, off"
                 :: "v"(lds_off), "v"(gptr) : "memory");
}
static __device__ __forceinline__ void wait_async0() {
    asm volatile("s_wait_asynccnt 0x0" ::: "memory");
}

#ifdef HAVE_TDM
// -- CDNA5 Tensor Data Mover: 2D bf16 tile global -> LDS, TENSORcnt ---------
// D# group0: count=1 | lds_addr | global_addr[56:0] | type=2
// D# group1: data_size=2B | tensor_dim0/1 | tile_dim0/1 | dim0_stride
static __device__ __forceinline__ void tdm_load_2d(
    unsigned lds_off, const void* gaddr,
    unsigned tensor_w, unsigned tensor_h,     // in elements / rows
    unsigned tile_w, unsigned tile_h,         // in elements / rows
    unsigned stride_w)                        // row stride in elements
{
    unsigned long long ga = (unsigned long long)gaddr;
    uv4 g0;
    g0[0] = 1u;                                        // 1 valid descriptor
    g0[1] = lds_off;                                   // LDS byte address
    g0[2] = (unsigned)ga;                              // global addr lo
    g0[3] = (unsigned)(ga >> 32) | (2u << 30);         // addr[56:32] | type=2
    iv8 g1;
    g1[0] = (int)(1u << 16);                           // data_size=1 (2 bytes)
    g1[1] = (int)((tensor_w & 0xFFFFu) << 16);         // tensor_dim0[15:0]
    g1[2] = (int)(((tensor_w >> 16) & 0xFFFFu) |
                  ((tensor_h & 0xFFFFu) << 16));       // dim0[31:16]|dim1[15:0]
    g1[3] = (int)(((tensor_h >> 16) & 0xFFFFu) |
                  (tile_w << 16));                     // dim1[31:16]|tile_dim0
    g1[4] = (int)tile_h;                               // tile_dim1 (tile_dim2=0)
    g1[5] = (int)stride_w;                             // dim0_stride[31:0]
    g1[6] = 0;                                         // stride hi / dim1_stride
    g1[7] = 0;
    iv4 z4 = {0, 0, 0, 0};
    iv8 z8 = {0, 0, 0, 0, 0, 0, 0, 0};
    __builtin_amdgcn_tensor_load_to_lds(g0, g1, z4, z4, z8, 0);
}
#endif

// ---------------------------------------------------------------------------
// Kernel 0: zero q/k scratch (pad lanes d in [80,96) must be exactly 0).
// ---------------------------------------------------------------------------
__global__ __launch_bounds__(256) void zero_kernel(uint4* __restrict__ p, long n4) {
    long i = (long)blockIdx.x * 256 + threadIdx.x;
    if (i < n4) { uint4 z; z.x = z.y = z.z = z.w = 0u; p[i] = z; }
}

// ---------------------------------------------------------------------------
// Kernel 1: qkv = hs @ w_qkv^T + b_qkv ; relayout to head-major bf16.
//   q,k : [BH][S][96]   vT : [BH][96][S]
// ---------------------------------------------------------------------------
__global__ __launch_bounds__(256) void qkv_gemm_kernel(
    const float* __restrict__ hs, const float* __restrict__ wqkv,
    const float* __restrict__ bqkv,
    unsigned short* __restrict__ q, unsigned short* __restrict__ k,
    unsigned short* __restrict__ vT)
{
    __shared__ unsigned short Asm[128][32];   // 8 KB, bf16 A tile
    __shared__ unsigned short Bsm[128][32];   // 8 KB, bf16 B tile
    const int tid = threadIdx.x;
    const int wave = tid >> 5, lane = tid & 31;
    const int lane16 = lane & 15, laneHalf = lane >> 4;
    const int m0 = blockIdx.y * 128, n0 = blockIdx.x * 128;
    const int wm = (wave & 1) * 64, wn = (wave >> 1) * 32;

    v8f acc[4][2] = {};

    for (int kk = 0; kk < HID; kk += 32) {
        __syncthreads();
        // batch all 8 tile loads, then convert+store (1 wait, 16 perms)
        float4 va[4], vb[4];
        #pragma unroll
        for (int i = 0; i < 4; ++i) {
            int f = tid + i * 256;           // 0..1023 float4 ids
            int r = f >> 3, c4 = (f & 7) << 2;
            va[i] = *(const float4*)&hs[(long)(m0 + r) * HID + kk + c4];
            vb[i] = *(const float4*)&wqkv[(long)(n0 + r) * HID + kk + c4];
        }
        #pragma unroll
        for (int i = 0; i < 4; ++i) {
            int f = tid + i * 256;
            int r = f >> 3, c4 = (f & 7) << 2;
            uint2 pa; pa.x = pack_bf16(va[i].x, va[i].y);
                      pa.y = pack_bf16(va[i].z, va[i].w);
            *(uint2*)&Asm[r][c4] = pa;
            uint2 pb; pb.x = pack_bf16(vb[i].x, vb[i].y);
                      pb.y = pack_bf16(vb[i].z, vb[i].w);
            *(uint2*)&Bsm[r][c4] = pb;
        }
        __syncthreads();

        v8u af[4], bfr[2];
        #pragma unroll
        for (int sm = 0; sm < 4; ++sm) {     // A frag: k chunks 0..7 & 16..23
            const unsigned short* p = &Asm[wm + sm * 16 + lane16][0] + laneHalf * 8;
            af[sm] = mk8(*(const uint4*)p, *(const uint4*)(p + 16));
        }
        #pragma unroll
        for (int nt = 0; nt < 2; ++nt) {     // B frag: k chunks 0..7 & 16..23
            const unsigned short* p = &Bsm[wn + nt * 16 + lane16][0] + laneHalf * 8;
            bfr[nt] = mk8(*(const uint4*)p, *(const uint4*)(p + 16));
        }
        #pragma unroll
        for (int sm = 0; sm < 4; ++sm)
            #pragma unroll
            for (int nt = 0; nt < 2; ++nt)
                acc[sm][nt] = wmma_bf16(af[sm], bfr[nt], acc[sm][nt]);
    }

    // epilogue: bias + relayout.  g (and which/h/d/bias) depends only on nt;
    // `which` is uniform per workgroup (2560 % 128 == 0) -> uniform branch.
    #pragma unroll
    for (int nt = 0; nt < 2; ++nt) {
        const int g = n0 + wn + nt * 16 + lane16;
        const float bias = bqkv[g];
        const int which = g / HID;            // 0=q 1=k 2=v
        const int gq = g - which * HID;
        const int h = gq / HD, d = gq - h * HD;
        unsigned short* const dstqk = (which == 0) ? q : k;
        #pragma unroll
        for (int sm = 0; sm < 4; ++sm)
            #pragma unroll
            for (int i = 0; i < 8; ++i) {
                int t = m0 + wm + sm * 16 + i + 8 * laneHalf;
                int b = t >> 10, s = t & (SEQ - 1);
                long bh = (long)(b * NH + h);
                unsigned short bv = f2bf(acc[sm][nt][i] + bias);
                if (which < 2) dstqk[(bh * SEQ + s) * DPAD + d] = bv;
                else           vT[(bh * DPAD + d) * SEQ + s] = bv;
            }
    }
}

// ---------------------------------------------------------------------------
// Kernel 2: in-place partial rotary on q and k  (pairs d, d+16 for d<16).
// ---------------------------------------------------------------------------
__global__ __launch_bounds__(256) void rope_kernel(
    unsigned short* __restrict__ q, unsigned short* __restrict__ k,
    const float* __restrict__ cosT, const float* __restrict__ sinT)
{
    int idx = blockIdx.x * 256 + threadIdx.x;       // 2 * 128 * 1024 * 16 threads
    int j   = idx & 15;
    int s   = (idx >> 4) & (SEQ - 1);
    int bh  = (idx >> 14) & (NB * NH - 1);
    int which = idx >> 21;
    unsigned short* buf = which ? k : q;
    int b = bh >> 5;
    int t = b * SEQ + s;
    float c  = cosT[t * RHALF + j];
    float sn = sinT[t * RHALF + j];
    unsigned short* p = buf + ((long)bh * SEQ + s) * DPAD;
    float x1 = bf2f(p[j]), x2 = bf2f(p[j + RHALF]);
    p[j]          = f2bf(x1 * c - x2 * sn);
    p[j + RHALF]  = f2bf(x2 * c + x1 * sn);
}

// ---------------------------------------------------------------------------
// Kernel 3: causal flash attention. WG = one (bh, 128-row q tile), 8 waves,
// wave owns 16 q rows. Streaming softmax over 32-key blocks.
// K/V tiles: double-buffered TDM pipeline (wave 0 issues next block's pair
// after the top barrier, waits tensorcnt<=2 so the previous pair is resident).
// ---------------------------------------------------------------------------
__global__ __launch_bounds__(256) void attn_kernel(
    const unsigned short* __restrict__ q, const unsigned short* __restrict__ k,
    const unsigned short* __restrict__ vT, unsigned short* __restrict__ attn)
{
    __shared__ unsigned short Ks[2][32][96];    // keys x d   (2 x 6 KB)
    __shared__ unsigned short Vs[2][96][32];    // d x keys   (2 x 6 KB)
    __shared__ unsigned short Ps[8][16][32];    // wave-private P tiles (8 KB)

    const int tid = threadIdx.x, wave = tid >> 5, lane = tid & 31;
    const int lane16 = lane & 15, laneHalf = lane >> 4;
    const int bh = blockIdx.y;
    const int b  = bh >> 5, h = bh & 31;
    const int q0 = blockIdx.x * 128;
    const int qrow0 = q0 + wave * 16;

    // Q fragments (A layout), 3 K-steps over d=0..95 (pad zeroed)
    v8u qf[3];
    {
        const unsigned short* qrow = q + ((long)bh * SEQ + qrow0 + lane16) * DPAD;
        #pragma unroll
        for (int s3 = 0; s3 < 3; ++s3) {
            const unsigned short* p = qrow + s3 * 32 + laneHalf * 8;
            qf[s3] = mk8(*(const uint4*)p, *(const uint4*)(p + 16));
        }
    }

    v8f o[5] = {};
    float Mrun[8], Lrun[8];
    #pragma unroll
    for (int i = 0; i < 8; ++i) { Mrun[i] = -1e30f; Lrun[i] = 0.0f; }

    const int kend = q0 + 128;

#ifdef HAVE_TDM
    if (wave == 0) {    // preload first K/V pair into buffer 0
        tdm_load_2d((unsigned)(unsigned long long)&Ks[0][0][0],
                    &k[(long)bh * SEQ * DPAD], DPAD, SEQ, DPAD, 32, DPAD);
        tdm_load_2d((unsigned)(unsigned long long)&Vs[0][0][0],
                    &vT[(long)bh * DPAD * SEQ], SEQ, DPAD, 32, DPAD, SEQ);
    }
#endif

    for (int kb = 0; kb < kend; kb += 32) {
        const int p2 = (kb >> 5) & 1;
        __syncthreads();   // prev compute done: safe to refill the other buffer
#ifdef HAVE_TDM
        if (wave == 0) {
            if (kb + 32 < kend) {
                tdm_load_2d((unsigned)(unsigned long long)&Ks[p2 ^ 1][0][0],
                            &k[((long)bh * SEQ + kb + 32) * DPAD],
                            DPAD, SEQ, DPAD, 32, DPAD);
                tdm_load_2d((unsigned)(unsigned long long)&Vs[p2 ^ 1][0][0],
                            &vT[(long)bh * DPAD * SEQ + kb + 32],
                            SEQ, DPAD, 32, DPAD, SEQ);
                __builtin_amdgcn_s_wait_tensorcnt(2);  // current pair resident
            } else {
                __builtin_amdgcn_s_wait_tensorcnt(0);  // last block: drain all
            }
        }
#else
        // fallback: per-lane async copies into buffer p2
        for (int idx = tid; idx < 384; idx += 256) {
            int r = idx / 12, c = idx - r * 12;
            async_b128(&Ks[p2][r][c * 8],
                       &k[((long)bh * SEQ + kb + r) * DPAD + c * 8]);
        }
        for (int idx = tid; idx < 384; idx += 256) {
            int d = idx >> 2, c = idx & 3;
            async_b128(&Vs[p2][d][c * 8],
                       &vT[((long)bh * DPAD + d) * SEQ + kb + c * 8]);
        }
        wait_async0();
#endif
        __syncthreads();

        // scores = Q @ K^T : 2 N-tiles x 3 K-steps
        v8f sc[2] = {};
        #pragma unroll
        for (int nt = 0; nt < 2; ++nt)
            #pragma unroll
            for (int s3 = 0; s3 < 3; ++s3) {
                const unsigned short* p = &Ks[p2][nt * 16 + lane16][s3 * 32] + laneHalf * 8;
                v8u bfr = mk8(*(const uint4*)p, *(const uint4*)(p + 16));
                sc[nt] = wmma_bf16(qf[s3], bfr, sc[nt]);
            }

        // scale + causal mask
        #pragma unroll
        for (int nt = 0; nt < 2; ++nt)
            #pragma unroll
            for (int i = 0; i < 8; ++i) {
                int mg = qrow0 + i + 8 * laneHalf;
                int kg2 = kb + nt * 16 + lane16;
                float v = sc[nt][i] * SCALE;
                sc[nt][i] = (kg2 <= mg) ? v : -1e30f;
            }

        // online softmax (lanes L and L+16 hold different rows -> xor 1,2,4,8)
        #pragma unroll
        for (int i = 0; i < 8; ++i) {
            float rmax = fmaxf(sc[0][i], sc[1][i]);
            rmax = fmaxf(rmax, __shfl_xor(rmax, 1, 32));
            rmax = fmaxf(rmax, __shfl_xor(rmax, 2, 32));
            rmax = fmaxf(rmax, __shfl_xor(rmax, 4, 32));
            rmax = fmaxf(rmax, __shfl_xor(rmax, 8, 32));
            float Mn = fmaxf(Mrun[i], rmax);
            float alpha = __expf(Mrun[i] - Mn);
            float p0 = __expf(sc[0][i] - Mn);
            float p1 = __expf(sc[1][i] - Mn);
            sc[0][i] = p0; sc[1][i] = p1;
            float rs = p0 + p1;
            rs += __shfl_xor(rs, 1, 32);
            rs += __shfl_xor(rs, 2, 32);
            rs += __shfl_xor(rs, 4, 32);
            rs += __shfl_xor(rs, 8, 32);
            Lrun[i] = Lrun[i] * alpha + rs;
            Mrun[i] = Mn;
            #pragma unroll
            for (int dt = 0; dt < 5; ++dt) o[dt][i] *= alpha;
        }

        // C-layout -> A-layout for P via wave-private LDS
        #pragma unroll
        for (int i = 0; i < 8; ++i) {
            Ps[wave][i + 8 * laneHalf][lane16]      = f2bf(sc[0][i]);
            Ps[wave][i + 8 * laneHalf][16 + lane16] = f2bf(sc[1][i]);
        }
        asm volatile("s_wait_dscnt 0x0" ::: "memory");
        v8u pf;
        {
            const unsigned short* p = &Ps[wave][lane16][0] + laneHalf * 8;
            pf = mk8(*(const uint4*)p, *(const uint4*)(p + 16));
        }

        // O += P @ V : 5 d-tiles, one K=32 step each
        #pragma unroll
        for (int dt = 0; dt < 5; ++dt) {
            const unsigned short* p = &Vs[p2][dt * 16 + lane16][0] + laneHalf * 8;
            v8u vf = mk8(*(const uint4*)p, *(const uint4*)(p + 16));
            o[dt] = wmma_bf16(pf, vf, o[dt]);
        }
    }

    // normalize + write token-major bf16 attn output [T][HID]
    #pragma unroll
    for (int i = 0; i < 8; ++i) {
        float inv = 1.0f / Lrun[i];
        int srow = qrow0 + i + 8 * laneHalf;
        long t = (long)b * SEQ + srow;
        #pragma unroll
        for (int dt = 0; dt < 5; ++dt)
            attn[t * HID + h * HD + dt * 16 + lane16] = f2bf(o[dt][i] * inv);
    }
}

// ---------------------------------------------------------------------------
// Kernel 4: out = attn @ w_dense^T + b_dense   (f32 output)
// A tile (bf16) staged via TDM (async fallback); W converted in registers.
// ---------------------------------------------------------------------------
__global__ __launch_bounds__(256) void dense_gemm_kernel(
    const unsigned short* __restrict__ A, const float* __restrict__ W,
    const float* __restrict__ bias, float* __restrict__ out)
{
    __shared__ unsigned short Asm[128][32];
    __shared__ unsigned short Bsm[128][32];
    const int tid = threadIdx.x;
    const int wave = tid >> 5, lane = tid & 31;
    const int lane16 = lane & 15, laneHalf = lane >> 4;
    const int m0 = blockIdx.y * 128, n0 = blockIdx.x * 128;
    const int wm = (wave & 1) * 64, wn = (wave >> 1) * 32;

    v8f acc[4][2] = {};

    for (int kk = 0; kk < HID; kk += 32) {
        __syncthreads();
#ifdef HAVE_TDM
        if (wave == 0)     // one TDM op fetches the whole 128x32 bf16 A tile
            tdm_load_2d((unsigned)(unsigned long long)&Asm[0][0],
                        &A[(long)m0 * HID + kk], HID, T_TOK, 32, 128, HID);
#else
        #pragma unroll
        for (int i = 0; i < 2; ++i) {
            int idx = tid + i * 256;
            int r = idx >> 2, c = idx & 3;
            async_b128(&Asm[r][c * 8], &A[(long)(m0 + r) * HID + kk + c * 8]);
        }
#endif
        // stage + convert W f32 -> bf16 (overlaps the A-tile DMA)
        float4 vb[4];
        #pragma unroll
        for (int i = 0; i < 4; ++i) {
            int f = tid + i * 256;
            int r = f >> 3, c4 = (f & 7) << 2;
            vb[i] = *(const float4*)&W[(long)(n0 + r) * HID + kk + c4];
        }
        #pragma unroll
        for (int i = 0; i < 4; ++i) {
            int f = tid + i * 256;
            int r = f >> 3, c4 = (f & 7) << 2;
            uint2 pb; pb.x = pack_bf16(vb[i].x, vb[i].y);
                      pb.y = pack_bf16(vb[i].z, vb[i].w);
            *(uint2*)&Bsm[r][c4] = pb;
        }
#ifdef HAVE_TDM
        if (wave == 0) __builtin_amdgcn_s_wait_tensorcnt(0);
#else
        wait_async0();
#endif
        __syncthreads();

        v8u af[4], bfr[2];
        #pragma unroll
        for (int sm = 0; sm < 4; ++sm) {
            const unsigned short* p = &Asm[wm + sm * 16 + lane16][0] + laneHalf * 8;
            af[sm] = mk8(*(const uint4*)p, *(const uint4*)(p + 16));
        }
        #pragma unroll
        for (int nt = 0; nt < 2; ++nt) {
            const unsigned short* p = &Bsm[wn + nt * 16 + lane16][0] + laneHalf * 8;
            bfr[nt] = mk8(*(const uint4*)p, *(const uint4*)(p + 16));
        }
        #pragma unroll
        for (int sm = 0; sm < 4; ++sm)
            #pragma unroll
            for (int nt = 0; nt < 2; ++nt)
                acc[sm][nt] = wmma_bf16(af[sm], bfr[nt], acc[sm][nt]);
    }

    // epilogue: bias load hoisted per nt (g independent of sm/i)
    #pragma unroll
    for (int nt = 0; nt < 2; ++nt) {
        const int g = n0 + wn + nt * 16 + lane16;
        const float bs = bias[g];
        #pragma unroll
        for (int sm = 0; sm < 4; ++sm)
            #pragma unroll
            for (int i = 0; i < 8; ++i) {
                int t = m0 + wm + sm * 16 + i + 8 * laneHalf;
                out[(long)t * HID + g] = acc[sm][nt][i] + bs;
            }
    }
}

// ---------------------------------------------------------------------------
// launch
// ---------------------------------------------------------------------------
extern "C" void kernel_launch(void* const* d_in, const int* in_sizes, int n_in,
                              void* d_out, int out_size, void* d_ws, size_t ws_size,
                              hipStream_t stream)
{
    const float* hs     = (const float*)d_in[0];
    const float* cosT   = (const float*)d_in[1];
    const float* sinT   = (const float*)d_in[2];
    const float* wqkv   = (const float*)d_in[3];
    const float* bqkv   = (const float*)d_in[4];
    const float* wdense = (const float*)d_in[5];
    const float* bdense = (const float*)d_in[6];
    float* out = (float*)d_out;

    const long QKELEM = (long)NB * NH * SEQ * DPAD;     // 12,582,912 bf16 each
    unsigned short* q    = (unsigned short*)d_ws;
    unsigned short* kbuf = q + QKELEM;
    unsigned short* vT   = kbuf + QKELEM;
    unsigned short* attn = vT + QKELEM;

    // 0) zero q+k (pad lanes must be 0 for the padded-K WMMA steps)
    long n4 = (2 * QKELEM * 2) / 16;                    // uint4 count
    zero_kernel<<<(unsigned)((n4 + 255) / 256), 256, 0, stream>>>((uint4*)q, n4);

    // 1) QKV GEMM + bias + relayout
    qkv_gemm_kernel<<<dim3(NQKV / 128, T_TOK / 128), 256, 0, stream>>>(
        hs, wqkv, bqkv, q, kbuf, vT);

    // 2) partial RoPE in place on q, k
    rope_kernel<<<(2 * NB * NH * SEQ * RHALF) / 256, 256, 0, stream>>>(
        q, kbuf, cosT, sinT);

    // 3) causal flash attention
    attn_kernel<<<dim3(SEQ / 128, NB * NH), 256, 0, stream>>>(q, kbuf, vT, attn);

    // 4) dense output GEMM
    dense_gemm_kernel<<<dim3(HID / 128, T_TOK / 128), 256, 0, stream>>>(
        attn, wdense, bdense, out);
}
